// MultiQueryAttention_65876208386635
// MI455X (gfx1250) — matrix operand order repaired and checked by reading, not compiled, not run on hardware
//
#include <hip/hip_runtime.h>

// ---------------------------------------------------------------------------
// MQA forward for MI455X (gfx1250): v_wmma_f32_16x16x32_bf16 for all matmuls,
// double-buffered TDM (tensor_load_to_lds, TENSORcnt-overlapped) tile staging,
// DPP16 row reductions for the online softmax.
// ---------------------------------------------------------------------------

typedef __bf16 bf16;
typedef __attribute__((ext_vector_type(16))) __bf16 v16bf;
typedef __attribute__((ext_vector_type(8)))  __bf16 bf16x8;
typedef __attribute__((ext_vector_type(8)))  float  v8f;
typedef unsigned int u32;
typedef unsigned long long u64;
typedef __attribute__((ext_vector_type(4))) u32 u32x4;
typedef __attribute__((ext_vector_type(8))) u32 u32x8;

#define N_EMBED 2048
#define N_HEAD  16
#define HEAD_DIM 128
#define BATCH   2
#define SEQ     2048
#define QKV_N   2304   // (16+2)*128

static __device__ inline v8f zero8() {
    v8f z;
#pragma unroll
    for (int i = 0; i < 8; ++i) z[i] = 0.0f;
    return z;
}

// A-fragment (16x32, M = lane&15): elems 0..7 at K = 8*half, elems 8..15 at K = 16+8*half
static __device__ inline v16bf load_fragA(const bf16* rowptr, int half) {
    union { v16bf v; bf16x8 h[2]; } u;
    u.h[0] = *(const bf16x8*)(rowptr + 8 * half);
    u.h[1] = *(const bf16x8*)(rowptr + 16 + 8 * half);
    return u.v;
}
// B-fragment (32x16, N = lane&15, data stored [N][K]): elems 0..7 at K = 8*half, elems 8..15 at K = 16+8*half
static __device__ inline v16bf load_fragB(const bf16* rowptr, int half) {
    union { v16bf v; bf16x8 h[2]; } u;
    u.h[0] = *(const bf16x8*)(rowptr + 8 * half);
    u.h[1] = *(const bf16x8*)(rowptr + 16 + 8 * half);
    return u.v;
}

static __device__ inline v8f wmma_bf16(v16bf a, v16bf b, v8f c) {
    return __builtin_amdgcn_wmma_f32_16x16x32_bf16(false, a, false, b, (short)0, c,
                                                   false, false);
}

// ---------------- DPP16 reductions over each 16-lane row -------------------
template <int CTRL>
static __device__ inline float dpp_mov_f(float x) {
    return __builtin_bit_cast(float,
        __builtin_amdgcn_mov_dpp(__builtin_bit_cast(int, x), CTRL, 0xF, 0xF, true));
}
static __device__ inline float rowmax16(float x) {
    x = fmaxf(x, dpp_mov_f<0x0B1>(x));   // quad_perm [1,0,3,2]  (xor 1)
    x = fmaxf(x, dpp_mov_f<0x04E>(x));   // quad_perm [2,3,0,1]  (xor 2)
    x = fmaxf(x, dpp_mov_f<0x141>(x));   // row_half_mirror      (xor 7)
    x = fmaxf(x, dpp_mov_f<0x140>(x));   // row_mirror           (xor 15)
    return x;
}
static __device__ inline float rowsum16(float x) {
    x += dpp_mov_f<0x0B1>(x);
    x += dpp_mov_f<0x04E>(x);
    x += dpp_mov_f<0x141>(x);
    x += dpp_mov_f<0x140>(x);
    return x;
}

// ---------------- Tensor Data Mover: 2D tile -> LDS (async issue) ----------
// D# per CDNA5 ISA ch.8. 2-group asm form (2D tensor, VADDR2/3 = NULL).
// Issue only; completion via s_wait_tensorcnt + workgroup barrier.
static __device__ __attribute__((always_inline)) void tdm_issue_2d(
        u32 lds_off, const bf16* gbase, u32 tile_cols, u32 tile_rows,
        u32 row_len, u64 row_stride_elems, u32 tensor_rows) {
    u64 ga = (u64)(uintptr_t)gbase;
    u32x4 g0;
    g0[0] = 1u;                                            // count=1 (valid, user)
    g0[1] = lds_off;                                       // lds_addr
    g0[2] = (u32)ga;                                       // global_addr[31:0]
    g0[3] = ((u32)(ga >> 32) & 0x01FFFFFFu) | (2u << 30);  // ga[56:32] | type=2
    u32x8 g1;
    g1[0] = 1u << 16;                                      // data_size=1 (2 bytes)
    g1[1] = (row_len & 0xFFFFu) << 16;                     // tensor_dim0[15:0]
    g1[2] = ((row_len >> 16) & 0xFFFFu) | ((tensor_rows & 0xFFFFu) << 16);
    g1[3] = ((tensor_rows >> 16) & 0xFFFFu) | (tile_cols << 16);  // tile_dim0
    g1[4] = tile_rows;                                     // tile_dim1 (tile_dim2=0)
    g1[5] = (u32)row_stride_elems;                         // tensor_dim0_stride lo
    g1[6] = (u32)(row_stride_elems >> 32) & 0xFFFFu;       // stride hi (dim1_stride=0)
    g1[7] = 0u;
    asm volatile("tensor_load_to_lds %0, %1"
                 :: "s"(g0), "s"(g1) : "memory");
}
static __device__ __attribute__((always_inline)) void tdm_wait() {
    __builtin_amdgcn_s_wait_tensorcnt(0);
}

// ------------------------- convert fp32 -> bf16 ----------------------------
__global__ __launch_bounds__(256) void cvt_f32_bf16(const float* __restrict__ src,
                                                    bf16* __restrict__ dst, size_t n) {
    size_t i = (size_t)blockIdx.x * blockDim.x + threadIdx.x;
    if (i < n) dst[i] = (bf16)src[i];
}

// -------------------- transpose fp32 (RxC) -> bf16 (CxR) -------------------
__global__ __launch_bounds__(256) void transpose_f32_bf16(const float* __restrict__ src,
                                                          bf16* __restrict__ dst,
                                                          int R, int C) {
    __shared__ float tile[32][33];
    int c0 = blockIdx.x * 32, r0 = blockIdx.y * 32;
    int tx = threadIdx.x & 31, ty = threadIdx.x >> 5;  // 32 x 8
#pragma unroll
    for (int i = ty; i < 32; i += 8)
        tile[i][tx] = src[(size_t)(r0 + i) * C + c0 + tx];
    __syncthreads();
#pragma unroll
    for (int i = ty; i < 32; i += 8)
        dst[(size_t)(c0 + i) * R + r0 + tx] = (bf16)tile[tx][i];
}

// ----------------------------- WMMA GEMM -----------------------------------
// C(MxN) = A(MxK) * Bt^T, A row-major bf16, Bt = B^T (N x K) row-major bf16.
// Workgroup tile 128x256, 8 waves of 64x64 (16 accum tiles), K-stage 64,
// double-buffered TDM: DMA for stage n+1 runs under the WMMAs of stage n.
template <bool BF16OUT>
__global__ __launch_bounds__(256) void gemm_wmma(const bf16* __restrict__ A,
                                                 const bf16* __restrict__ Bt,
                                                 void* __restrict__ Cout,
                                                 int M, int N, int K) {
    __shared__ bf16 As[2][128][64];   // 2 x 16 KB
    __shared__ bf16 Bs[2][256][64];   // 2 x 32 KB

    const int t = threadIdx.x;
    const int wave = t >> 5, lane = t & 31;
    const int half = lane >> 4, ln = lane & 15;
    const int wm = (wave >> 2) * 64;   // 0 / 64
    const int wn = (wave & 3) * 64;    // 0 / 64 / 128 / 192

    const bf16* Arow = A + (size_t)blockIdx.x * 128 * K;
    const bf16* Brow = Bt + (size_t)blockIdx.y * 256 * K;

    v8f acc[4][4];
#pragma unroll
    for (int i = 0; i < 4; ++i)
#pragma unroll
        for (int j = 0; j < 4; ++j) acc[i][j] = zero8();

    // prologue: stage 0 in flight
    if (wave == 0)
        tdm_issue_2d((u32)(uintptr_t)&As[0][0][0], Arow, 64, 128, (u32)K, (u64)K, 128);
    else if (wave == 1)
        tdm_issue_2d((u32)(uintptr_t)&Bs[0][0][0], Brow, 64, 256, (u32)K, (u64)K, 256);

    int buf = 0;
    for (int k0 = 0; k0 < K; k0 += 64, buf ^= 1) {
        tdm_wait();            // current buffer's DMA complete (issuing waves)
        __syncthreads();       // publish tile; also: all waves done reading buf^1
        if (k0 + 64 < K) {     // prefetch next stage into the other buffer
            if (wave == 0)
                tdm_issue_2d((u32)(uintptr_t)&As[buf ^ 1][0][0], Arow + k0 + 64,
                             64, 128, (u32)K, (u64)K, 128);
            else if (wave == 1)
                tdm_issue_2d((u32)(uintptr_t)&Bs[buf ^ 1][0][0], Brow + k0 + 64,
                             64, 256, (u32)K, (u64)K, 256);
        }
#pragma unroll
        for (int kk = 0; kk < 64; kk += 32) {
            v16bf bfrag[4];
#pragma unroll
            for (int jn = 0; jn < 4; ++jn)
                bfrag[jn] = load_fragB(&Bs[buf][wn + jn * 16 + ln][kk], half);
#pragma unroll
            for (int im = 0; im < 4; ++im) {
                v16bf afrag = load_fragA(&As[buf][wm + im * 16 + ln][kk], half);
#pragma unroll
                for (int jn = 0; jn < 4; ++jn)
                    acc[im][jn] = wmma_bf16(afrag, bfrag[jn], acc[im][jn]);
            }
        }
    }

    // C layout: VGPR i -> row i + 8*half, col = lane&15
#pragma unroll
    for (int im = 0; im < 4; ++im)
#pragma unroll
        for (int jn = 0; jn < 4; ++jn)
#pragma unroll
            for (int i = 0; i < 8; ++i) {
                size_t row = (size_t)blockIdx.x * 128 + wm + im * 16 + i + 8 * half;
                size_t col = (size_t)blockIdx.y * 256 + wn + jn * 16 + ln;
                if (BF16OUT)
                    ((bf16*)Cout)[row * N + col] = (bf16)acc[im][jn][i];
                else
                    ((float*)Cout)[row * N + col] = acc[im][jn][i];
            }
    (void)M;
}

// --------------------------- RoPE + QKV split ------------------------------
// qkv (B*T, 2304) bf16 -> q (B,H,T,hd) *pre-scaled by 1/sqrt(hd)*, k (B,T,hd),
// vT (B,hd,T), all bf16. RoPE is linear, so folding the softmax scale into q
// is exact and removes 16 v_mul per tile from the flash inner loop.
__global__ __launch_bounds__(256) void rope_split(const bf16* __restrict__ qkv,
                                                  bf16* __restrict__ qb,
                                                  bf16* __restrict__ kb,
                                                  bf16* __restrict__ vT) {
    const int row = blockIdx.x;          // 0..4095
    const int b = row >> 11, tt = row & 2047;
    const bf16* src = qkv + (size_t)row * QKV_N;
    const float tf = (float)tt;
    const float scale = 0.08838834764831845f;   // 1/sqrt(128)

    for (int c = threadIdx.x; c < N_EMBED; c += blockDim.x) {
        int h = c >> 7, d = c & 127, j = d & 63;
        float inv = __powf(10000.0f, (-(4.0f * (float)j) - 1.0f) / 128.0f);
        float a = tf * inv;
        float ca = __cosf(a), sa = __sinf(a);
        float x = (float)src[c];
        float xr = (d < 64) ? -(float)src[c + 64] : (float)src[c - 64];
        qb[(((size_t)b * N_HEAD + h) * SEQ + tt) * HEAD_DIM + d] =
            (bf16)((x * ca + xr * sa) * scale);
    }
    for (int d = threadIdx.x; d < HEAD_DIM; d += blockDim.x) {
        int j = d & 63;
        float inv = __powf(10000.0f, (-(4.0f * (float)j) - 1.0f) / 128.0f);
        float a = tf * inv;
        float ca = __cosf(a), sa = __sinf(a);
        float x = (float)src[N_EMBED + d];
        float xr = (d < 64) ? -(float)src[N_EMBED + d + 64] : (float)src[N_EMBED + d - 64];
        kb[((size_t)b * SEQ + tt) * HEAD_DIM + d] = (bf16)(x * ca + xr * sa);
    }
    for (int d = threadIdx.x; d < HEAD_DIM; d += blockDim.x)
        vT[((size_t)b * HEAD_DIM + d) * SEQ + tt] = src[N_EMBED + HEAD_DIM + d];
}

// --------------------- Flash MQA (WMMA, shared K/V) ------------------------
// 8 waves/WG: each wave owns a 16-query tile of some head; all waves share the
// same batch's K/V LDS tiles (multi-query => K/V shared across all 16 heads).
// K/V tiles double-buffered via TDM so the DMA runs under the WMMA/softmax.
__global__ __launch_bounds__(256) void mqa_flash(const bf16* __restrict__ qb,
                                                 const bf16* __restrict__ kb,
                                                 const bf16* __restrict__ vT,
                                                 bf16* __restrict__ out) {
    __shared__ bf16 Ks[2][32][128];     // [key][hd]   -> B-frag for Q@K^T
    __shared__ bf16 Vs[2][128][32];     // [hd][key]   -> B-frag for P@V
    __shared__ bf16 Ps[8][16][32];      // per-wave P tile

    const int g = blockIdx.x;           // 0..511
    const int b = g >> 8;
    const int t = threadIdx.x;
    const int wave = t >> 5, lane = t & 31;
    const int half = lane >> 4, ln = lane & 15;
    const int ti = (g & 255) * 8 + wave;      // 0..2047
    const int h = ti >> 7;                    // head
    const int q0 = (ti & 127) * 16;           // query row base

    // Q fragments (16 x 128, four K-slices of 32); q pre-scaled by 1/sqrt(hd)
    const bf16* qbase = qb + (((size_t)b * N_HEAD + h) * SEQ + q0) * HEAD_DIM;
    v16bf qf[4];
#pragma unroll
    for (int kf = 0; kf < 4; ++kf)
        qf[kf] = load_fragA(qbase + (size_t)ln * HEAD_DIM + kf * 32, half);

    float m[8], l[8];
    v8f O[8];
#pragma unroll
    for (int i = 0; i < 8; ++i) { m[i] = -1e30f; l[i] = 0.0f; O[i] = zero8(); }

    const bf16* kbb = kb + (size_t)b * SEQ * HEAD_DIM;
    const bf16* vtb = vT + (size_t)b * HEAD_DIM * SEQ;

    // prologue: first K/V tiles in flight
    if (wave == 0)
        tdm_issue_2d((u32)(uintptr_t)&Ks[0][0][0], kbb, 128, 32, 128, (u64)HEAD_DIM, SEQ);
    else if (wave == 1)
        tdm_issue_2d((u32)(uintptr_t)&Vs[0][0][0], vtb, 32, 128, SEQ, (u64)SEQ, 128);

    int buf = 0;
    for (int kblk = 0; kblk < SEQ; kblk += 32, buf ^= 1) {
        tdm_wait();
        __syncthreads();
        if (kblk + 32 < SEQ) {
            if (wave == 0)
                tdm_issue_2d((u32)(uintptr_t)&Ks[buf ^ 1][0][0],
                             kbb + (size_t)(kblk + 32) * HEAD_DIM,
                             128, 32, 128, (u64)HEAD_DIM, SEQ);
            else if (wave == 1)
                tdm_issue_2d((u32)(uintptr_t)&Vs[buf ^ 1][0][0], vtb + kblk + 32,
                             32, 128, SEQ, (u64)SEQ, 128);
        }

        // S = Q K^T for 32 keys (two 16x16 C-tiles), K-dim = 128 chained
        v8f S[2];
#pragma unroll
        for (int nt = 0; nt < 2; ++nt) {
            v8f c = zero8();
#pragma unroll
            for (int kf = 0; kf < 4; ++kf)
                c = wmma_bf16(qf[kf],
                              load_fragB(&Ks[buf][nt * 16 + ln][kf * 32], half), c);
            S[nt] = c;
        }

        // online softmax: row i+8*half per reg i; 16 cols live in the lane's
        // 16-lane row -> pure-VALU DPP16 reductions (no ds_bpermute).
#pragma unroll
        for (int i = 0; i < 8; ++i) {
            float s0 = S[0][i], s1 = S[1][i];
            float mx = rowmax16(fmaxf(s0, s1));
            float mn = fmaxf(m[i], mx);
            float corr = __expf(m[i] - mn);
            float p0 = __expf(s0 - mn), p1 = __expf(s1 - mn);
            float rs = rowsum16(p0 + p1);
            l[i] = l[i] * corr + rs;
            m[i] = mn;
#pragma unroll
            for (int j = 0; j < 8; ++j) O[j][i] *= corr;
            Ps[wave][i + 8 * half][ln]      = (bf16)p0;
            Ps[wave][i + 8 * half][16 + ln] = (bf16)p1;
        }
        // wave-local LDS RAW fence (Ps is wave-private; no workgroup barrier needed)
        asm volatile("s_wait_dscnt 0x0" ::: "memory");

        v16bf pf = load_fragA(&Ps[wave][ln][0], half);
#pragma unroll
        for (int j = 0; j < 8; ++j)
            O[j] = wmma_bf16(pf, load_fragB(&Vs[buf][j * 16 + ln][0], half), O[j]);
    }

    bf16* obase = out + ((size_t)b * SEQ + q0) * N_EMBED + h * HEAD_DIM;
#pragma unroll
    for (int j = 0; j < 8; ++j)
#pragma unroll
        for (int i = 0; i < 8; ++i)
            obase[(size_t)(i + 8 * half) * N_EMBED + j * 16 + ln] = (bf16)(O[j][i] / l[i]);
}

// ---------------------------------------------------------------------------
extern "C" void kernel_launch(void* const* d_in, const int* in_sizes, int n_in,
                              void* d_out, int out_size, void* d_ws, size_t ws_size,
                              hipStream_t stream) {
    (void)in_sizes; (void)n_in; (void)out_size; (void)ws_size;
    const float* x      = (const float*)d_in[0];   // (B,T,E)
    const float* w_attn = (const float*)d_in[1];   // (E, 2304)
    const float* w_out  = (const float*)d_in[2];   // (E, E)
    float* outp = (float*)d_out;

    char* ws = (char*)d_ws;
    const size_t BT = (size_t)BATCH * SEQ;               // 4096
    bf16* xb   = (bf16*)(ws);                            // BT x E
    bf16* watT = (bf16*)(ws + 16777216);                 // 2304 x 2048
    bf16* woT  = (bf16*)(ws + 26214400);                 // 2048 x 2048
    bf16* qkv  = (bf16*)(ws + 34603008);                 // BT x 2304
    bf16* qb   = (bf16*)(ws + 53477376);                 // B,H,T,hd (pre-scaled)
    bf16* kbuf = (bf16*)(ws + 70254592);                 // B,T,hd
    bf16* vT   = (bf16*)(ws + 71303168);                 // B,hd,T
    bf16* ao   = (bf16*)(ws + 72351744);                 // BT x E

    // 1) fp32 -> bf16 converts / transposes
    {
        size_t n = BT * N_EMBED;
        cvt_f32_bf16<<<dim3((unsigned)((n + 255) / 256)), dim3(256), 0, stream>>>(x, xb, n);
        transpose_f32_bf16<<<dim3(QKV_N / 32, N_EMBED / 32), dim3(256), 0, stream>>>(
            w_attn, watT, N_EMBED, QKV_N);
        transpose_f32_bf16<<<dim3(N_EMBED / 32, N_EMBED / 32), dim3(256), 0, stream>>>(
            w_out, woT, N_EMBED, N_EMBED);
    }
    // 2) qkv = x @ w_attn  (bf16 out)
    gemm_wmma<true><<<dim3(BT / 128, QKV_N / 256), dim3(256), 0, stream>>>(
        xb, watT, qkv, (int)BT, QKV_N, N_EMBED);
    // 3) RoPE + split (+ fold softmax scale into q)
    rope_split<<<dim3((unsigned)BT), dim3(256), 0, stream>>>(qkv, qb, kbuf, vT);
    // 4) flash multi-query attention
    mqa_flash<<<dim3(512), dim3(256), 0, stream>>>(qb, kbuf, vT, ao);
    // 5) out = ao @ w_out  (fp32 out)
    gemm_wmma<false><<<dim3(BT / 128, N_EMBED / 256), dim3(256), 0, stream>>>(
        ao, woT, outp, (int)BT, N_EMBED, N_EMBED);
}
